// MultiStreamAttention_65893388255306
// MI455X (gfx1250) — compile-verified, matrix operand order checked by reading, not run on hardware
//
#include <hip/hip_runtime.h>

// ---------------------------------------------------------------------------
// MultiStreamAttention for MI455X (gfx1250), wave32 + WMMA bf16 path.
// GEMM inner loop: ping-pong register double-buffering (no copy rotation),
// WMMAs overlap in-flight b128 loads.
// ---------------------------------------------------------------------------

typedef __attribute__((ext_vector_type(16))) __bf16 v16bf;
typedef __attribute__((ext_vector_type(8)))  float  v8f;

struct alignas(16) B8 { __bf16 v[8]; };

// ---------------------------------------------------------------------------
// Elementwise fp32 -> bf16 conversion (one-time, off the GEMM critical path)
// ---------------------------------------------------------------------------
__global__ void cvt_bf16_kernel(const float* __restrict__ in,
                                __bf16* __restrict__ out, int n)
{
    int i = blockIdx.x * blockDim.x + threadIdx.x;
    if (i < n) out[i] = (__bf16)in[i];
}

// ---------------------------------------------------------------------------
// Fragment set for one 16x32 K-step of a 16x64 output tile.
// ---------------------------------------------------------------------------
struct Frags { B8 a0, a1, b0[4], b1[4]; };

__device__ __forceinline__ void load_frags(Frags& f, const __bf16* ap,
                                           const __bf16* bp0, int K, int k0)
{
    f.a0 = *(const B8*)(ap + k0);
    f.a1 = *(const B8*)(ap + k0 + 16);
    #pragma unroll
    for (int t = 0; t < 4; ++t) {
        const __bf16* bp = bp0 + (size_t)t * 16 * K + k0;
        f.b0[t] = *(const B8*)(bp);
        f.b1[t] = *(const B8*)(bp + 16);
    }
}

__device__ __forceinline__ void mma_frags(v8f (&c)[4], const Frags& f)
{
    v16bf a;
    #pragma unroll
    for (int j = 0; j < 8; ++j) { a[j] = f.a0.v[j]; a[8 + j] = f.a1.v[j]; }
    #pragma unroll
    for (int t = 0; t < 4; ++t) {
        v16bf b;
        #pragma unroll
        for (int j = 0; j < 8; ++j) { b[j] = f.b0[t].v[j]; b[8 + j] = f.b1[t].v[j]; }
        c[t] = __builtin_amdgcn_wmma_f32_16x16x32_bf16(
                   false, a, false, b, (short)0, c[t], false, false);
    }
}

// ---------------------------------------------------------------------------
// Generic  out[M,N] = A[M,K] @ W[N,K]^T (+bias), one wave per 16x64 tile.
// bf16 operands, fp32 accumulate.  Requires K a multiple of 64 (all call
// sites use K=384).  Ping-pong buffers fA/fB: no register copies.
// ---------------------------------------------------------------------------
__global__ void __launch_bounds__(256)
gemm_bf16_wmma(const __bf16* __restrict__ A, const __bf16* __restrict__ Wt,
               const float* __restrict__ bias,
               float* __restrict__ outF, __bf16* __restrict__ outB,
               int M, int N, int K)
{
    const int wave = (blockIdx.x * blockDim.x + threadIdx.x) >> 5;
    const int lane = threadIdx.x & 31;
    const int mtiles = M >> 4;
    const int T = mtiles * (N >> 6);
    if (wave >= T) return;                 // wave-uniform: EXEC stays all-ones
    const int m0 = (wave % mtiles) << 4;
    const int n0 = (wave / mtiles) << 6;
    const int ln15 = lane & 15;
    const int kofl = (lane >> 4) << 3;     // K sub-base 0/8 per half-wave

    const __bf16* ap  = A  + (size_t)(m0 + ln15) * K + kofl;
    const __bf16* bp0 = Wt + (size_t)(n0 + ln15) * K + kofl;

    const v8f vz = {0.f,0.f,0.f,0.f,0.f,0.f,0.f,0.f};
    v8f c[4] = {vz, vz, vz, vz};

    Frags fA, fB;
    load_frags(fA, ap, bp0, K, 0);                      // step 0 in flight

    int k0 = 32;
    for (; k0 + 32 < K; k0 += 64) {
        load_frags(fB, ap, bp0, K, k0);                 // step s   -> fB
        __builtin_prefetch(ap + k0 + 64, 0, 3);         // global_prefetch_b8
        mma_frags(c, fA);                               // step s-1
        load_frags(fA, ap, bp0, K, k0 + 32);            // step s+1 -> fA
        mma_frags(c, fB);                               // step s
    }
    load_frags(fB, ap, bp0, K, K - 32);                 // last step
    mma_frags(c, fA);
    mma_frags(c, fB);

    // ---- store (C/D layout: vgpr i -> row m0+i (+8 for upper half)) -----
    const int rbase = m0 + ((lane >> 4) << 3);
    #pragma unroll
    for (int t = 0; t < 4; ++t) {
        const int col = n0 + t*16 + ln15;
        const float bv = bias ? bias[col] : 0.f;
        #pragma unroll
        for (int i = 0; i < 8; ++i) {
            const size_t off = (size_t)(rbase + i) * N + col;
            const float v = c[t][i] + bv;
            if (outB) outB[off] = (__bf16)v;
            else      outF[off] = v;
        }
    }
}

// ---------------------------------------------------------------------------
// 4x4 average pool: x(B,64,64,384) f32 -> xc(B,256,384) bf16
// ---------------------------------------------------------------------------
__global__ void avgpool_kernel(const float* __restrict__ x, __bf16* __restrict__ xc)
{
    int i = blockIdx.x * blockDim.x + threadIdx.x;
    if (i >= 8*256*384) return;
    int c = i % 384, n = (i/384) % 256, b = i / (384*256);
    int hc = n >> 4, wc = n & 15;
    float s = 0.f;
    #pragma unroll
    for (int dy = 0; dy < 4; ++dy)
        #pragma unroll
        for (int dx = 0; dx < 4; ++dx)
            s += x[(((size_t)b*64 + hc*4 + dy)*64 + wc*4 + dx)*384 + c];
    xc[i] = (__bf16)(s * (1.f/16.f));
}

// ---------------------------------------------------------------------------
// Coarse attention: per (b,h,q) thread, online softmax over 256 keys,
// top-4 index tracking, xo(B,Hc,Wc,C) output.  qkvc layout (B*256, 1152) f32.
// ---------------------------------------------------------------------------
__global__ void coarse_attn_kernel(const float* __restrict__ qkvc,
                                   float* __restrict__ xo, int* __restrict__ topk)
{
    int t = blockIdx.x * blockDim.x + threadIdx.x;
    if (t >= 8*8*256) return;
    int nq = t % 256, h = (t/256) % 8, b = t / (256*8);
    const float scale = rsqrtf(48.f);

    const float* qr = qkvc + (size_t)(b*256 + nq)*1152 + h*48;
    float q[48], acc[48];
    #pragma unroll
    for (int j = 0; j < 48; ++j) { q[j] = qr[j] * scale; acc[j] = 0.f; }

    float m = -1e30f, l = 0.f;
    float tv[4] = {-1e30f,-1e30f,-1e30f,-1e30f};
    int   ti[4] = {0,0,0,0};

    for (int k = 0; k < 256; ++k) {
        const float* kr = qkvc + (size_t)(b*256 + k)*1152 + 384 + h*48;
        float s = 0.f;
        #pragma unroll
        for (int j = 0; j < 48; ++j) s += q[j] * kr[j];
        if (s > tv[3]) {                       // insertion into sorted top-4
            int p = 3;
            while (p > 0 && s > tv[p-1]) { tv[p] = tv[p-1]; ti[p] = ti[p-1]; --p; }
            tv[p] = s; ti[p] = k;
        }
        const float* vr = qkvc + (size_t)(b*256 + k)*1152 + 768 + h*48;
        float nm = fmaxf(m, s);
        float cm = __expf(m - nm), cs = __expf(s - nm);
        l = l*cm + cs;
        #pragma unroll
        for (int j = 0; j < 48; ++j) acc[j] = acc[j]*cm + cs*vr[j];
        m = nm;
    }
    float inv = 1.f / l;
    float* xr = xo + (size_t)(b*256 + nq)*384 + h*48;
    #pragma unroll
    for (int j = 0; j < 48; ++j) xr[j] = acc[j] * inv;
    int* ir = topk + (size_t)((b*8 + h)*256 + nq)*4;
    #pragma unroll
    for (int j = 0; j < 4; ++j) ir[j] = ti[j];
}

// ---------------------------------------------------------------------------
// Windowed gathered attention: one wave per (b,h,window).
//   S(16x64) = Qw(16x48) . Kg(64x48)^T  -> softmax -> yo = P(16x64) . Vg(64x48)
// d=48 padded to K=64 with zeroed bf16 fragments.  P staged through LDS (bf16)
// to convert the C/D layout into the A-matrix layout (ds_*_b128 traffic).
// qkvf: (B*4096, 1152) bf16.  yo: (B,64,64,384) f32.
// ---------------------------------------------------------------------------
__global__ void __launch_bounds__(256)
win_attn_kernel(const __bf16* __restrict__ qb,
                const int* __restrict__ topk, float* __restrict__ yo)
{
    __shared__ int    s_pix[8][64];
    __shared__ __bf16 s_P[8][16*64];

    const int wslot = threadIdx.x >> 5;
    const int lane  = threadIdx.x & 31;
    const int gw    = blockIdx.x * 8 + wslot;       // (b,h,nc) flat, exact grid
    const int nc = gw % 256, h = (gw/256) % 8, b = gw / (256*8);
    const int hc = nc >> 4, wc = nc & 15;
    const int lf   = (lane >> 4) << 3;              // 0 / 8 half-wave offset
    const int ln15 = lane & 15;
    const __bf16 bz = (__bf16)0.f;
    const v8f vz = {0.f,0.f,0.f,0.f,0.f,0.f,0.f,0.f};

    // --- gathered key pixel row bases (elements into qkvf) ----------------
    const int* ir = topk + (size_t)((b*8 + h)*256 + nc)*4;
    for (int k = lane; k < 64; k += 32) {
        int w = ir[k >> 4];
        int p = k & 15;
        int py = ((w >> 4) << 2) + (p >> 2);
        int px = ((w & 15) << 2) + (p & 3);
        s_pix[wslot][k] = (b*4096 + py*64 + px) * 1152;
    }
    __syncthreads();

    // --- Q fragments (A-matrix, 16x32 bf16, two K-slices, pad d 48->64) ---
    v16bf aq1, aq2;
    {
        int qrow = ln15;
        int py = (hc << 2) + (qrow >> 2), px = (wc << 2) + (qrow & 3);
        int base = (b*4096 + py*64 + px)*1152 + h*48;
        B8 t0 = *(const B8*)(qb + base + lf);
        B8 t1 = *(const B8*)(qb + base + lf + 16);
        B8 t2 = *(const B8*)(qb + base + lf + 32);
        #pragma unroll
        for (int j = 0; j < 8; ++j) {
            aq1[j] = t0.v[j]; aq1[8+j] = t1.v[j];
            aq2[j] = t2.v[j]; aq2[8+j] = bz;        // d 48..63 = 0
        }
    }

    // --- S = Q . Kg^T : 4 key tiles x 2 K-steps = 8 WMMAs -----------------
    v8f s[4] = {vz, vz, vz, vz};
    #pragma unroll
    for (int t = 0; t < 4; ++t) {
        int key  = t*16 + ln15;                     // this lane's B column
        int base = s_pix[wslot][key] + 384 + h*48;  // K section
        B8 k0 = *(const B8*)(qb + base + lf);
        B8 k1 = *(const B8*)(qb + base + lf + 16);
        B8 k2 = *(const B8*)(qb + base + lf + 32);
        v16bf b1, b2;
        #pragma unroll
        for (int j = 0; j < 8; ++j) {
            b1[j] = k0.v[j]; b1[8+j] = k1.v[j];
            b2[j] = k2.v[j]; b2[8+j] = bz;
        }
        s[t] = __builtin_amdgcn_wmma_f32_16x16x32_bf16(false, aq1, false, b1,
                                                       (short)0, s[t], false, false);
        s[t] = __builtin_amdgcn_wmma_f32_16x16x32_bf16(false, aq2, false, b2,
                                                       (short)0, s[t], false, false);
    }

    // --- softmax over 64 keys; rows live in 16-lane groups ----------------
    const float scale = rsqrtf(48.f);
    __bf16* P = s_P[wslot];
    #pragma unroll
    for (int i = 0; i < 8; ++i) {
        float m = -1e30f;
        #pragma unroll
        for (int t = 0; t < 4; ++t) { s[t][i] *= scale; m = fmaxf(m, s[t][i]); }
        for (int msk = 1; msk < 16; msk <<= 1) m = fmaxf(m, __shfl_xor(m, msk, 32));
        float e[4], l = 0.f;
        #pragma unroll
        for (int t = 0; t < 4; ++t) { e[t] = __expf(s[t][i] - m); l += e[t]; }
        for (int msk = 1; msk < 16; msk <<= 1) l += __shfl_xor(l, msk, 32);
        float inv = 1.f / l;
        int qrow = i + lf;                          // C/D row for this vgpr
        #pragma unroll
        for (int t = 0; t < 4; ++t)
            P[qrow*64 + t*16 + ln15] = (__bf16)(e[t] * inv);
    }
    __syncthreads();

    // --- reload P in A-matrix layout from LDS (ds_load_b128) --------------
    v16bf pa1, pa2;
    {
        int qrow = ln15;
        B8 p0 = *(const B8*)(P + qrow*64 + lf);
        B8 p1 = *(const B8*)(P + qrow*64 + lf + 16);
        B8 p2 = *(const B8*)(P + qrow*64 + lf + 32);
        B8 p3 = *(const B8*)(P + qrow*64 + lf + 48);
        #pragma unroll
        for (int j = 0; j < 8; ++j) {
            pa1[j] = p0.v[j]; pa1[8+j] = p1.v[j];
            pa2[j] = p2.v[j]; pa2[8+j] = p3.v[j];
        }
    }

    // --- yo = P . Vg : 3 d tiles x 2 K-steps = 6 WMMAs --------------------
    v8f o[3] = {vz, vz, vz};
    const int vsec = 768 + h*48;
    #pragma unroll
    for (int t = 0; t < 3; ++t) {
        int n = t*16 + ln15;                        // this lane's d column
        v16bf b1, b2;
        #pragma unroll
        for (int j = 0; j < 8; ++j) {
            b1[j]   = qb[s_pix[wslot][lf + j]       + vsec + n];
            b1[8+j] = qb[s_pix[wslot][lf + 16 + j]  + vsec + n];
            b2[j]   = qb[s_pix[wslot][lf + 32 + j]  + vsec + n];
            b2[8+j] = qb[s_pix[wslot][lf + 48 + j]  + vsec + n];
        }
        o[t] = __builtin_amdgcn_wmma_f32_16x16x32_bf16(false, pa1, false, b1,
                                                       (short)0, o[t], false, false);
        o[t] = __builtin_amdgcn_wmma_f32_16x16x32_bf16(false, pa2, false, b2,
                                                       (short)0, o[t], false, false);
    }

    // --- un-window and store ---------------------------------------------
    #pragma unroll
    for (int t = 0; t < 3; ++t) {
        int dcol = t*16 + ln15;
        #pragma unroll
        for (int i = 0; i < 8; ++i) {
            int qrow = i + lf;
            int py = (hc << 2) + (qrow >> 2), px = (wc << 2) + (qrow & 3);
            yo[((size_t)b*4096 + py*64 + px)*384 + h*48 + dcol] = o[t][i];
        }
    }
}

// ---------------------------------------------------------------------------
// t1 = 0.5*dwconv3x3(upsample4(xo)) + 0.5*yo   (in-place into yo buffer)
// ---------------------------------------------------------------------------
__global__ void upconv_mix_kernel(const float* __restrict__ xo,
                                  const float* __restrict__ dwx_w,
                                  const float* __restrict__ dwx_b,
                                  float* __restrict__ yo_t1)
{
    int i = blockIdx.x * blockDim.x + threadIdx.x;
    if (i >= 8*64*64*384) return;
    int c = i % 384, x = (i/384) & 63, y = (i/(384*64)) & 63, b = i / (384*4096);
    float s = dwx_b[c];
    #pragma unroll
    for (int dy = -1; dy <= 1; ++dy) {
        int yy = y + dy; if ((unsigned)yy >= 64u) continue;
        #pragma unroll
        for (int dx = -1; dx <= 1; ++dx) {
            int xx = x + dx; if ((unsigned)xx >= 64u) continue;
            float v = xo[(((size_t)b*16 + (yy >> 2))*16 + (xx >> 2))*384 + c];
            s += v * dwx_w[c*9 + (dy+1)*3 + (dx+1)];
        }
    }
    yo_t1[i] = 0.5f*s + 0.5f*yo_t1[i];
}

// ---------------------------------------------------------------------------
// z2 = dwconv3x3(t1) + dw_b   (bf16 output: feeds final WMMA GEMM)
// ---------------------------------------------------------------------------
__global__ void dwconv_kernel(const float* __restrict__ t1,
                              const float* __restrict__ dw_w,
                              const float* __restrict__ dw_b,
                              __bf16* __restrict__ z2)
{
    int i = blockIdx.x * blockDim.x + threadIdx.x;
    if (i >= 8*64*64*384) return;
    int c = i % 384, x = (i/384) & 63, y = (i/(384*64)) & 63, b = i / (384*4096);
    float s = dw_b[c];
    #pragma unroll
    for (int dy = -1; dy <= 1; ++dy) {
        int yy = y + dy; if ((unsigned)yy >= 64u) continue;
        #pragma unroll
        for (int dx = -1; dx <= 1; ++dx) {
            int xx = x + dx; if ((unsigned)xx >= 64u) continue;
            s += t1[(((size_t)b*64 + yy)*64 + xx)*384 + c] * dw_w[c*9 + (dy+1)*3 + (dx+1)];
        }
    }
    z2[i] = (__bf16)s;
}

// ---------------------------------------------------------------------------
// Launch sequence
// ---------------------------------------------------------------------------
extern "C" void kernel_launch(void* const* d_in, const int* in_sizes, int n_in,
                              void* d_out, int out_size, void* d_ws, size_t ws_size,
                              hipStream_t stream)
{
    const float* x     = (const float*)d_in[0];
    const float* qkv_w = (const float*)d_in[1];
    const float* dwx_w = (const float*)d_in[2];
    const float* dwx_b = (const float*)d_in[3];
    const float* dw_w  = (const float*)d_in[4];
    const float* dw_b  = (const float*)d_in[5];
    const float* pw_w  = (const float*)d_in[6];
    const float* pw_b  = (const float*)d_in[7];

    char* ws = (char*)d_ws;
    __bf16* xc_bf  = (__bf16*)(ws + 0);              //  1.5 MB
    float*  qkvc   = (float*) (ws + 1572864);        //  9.4 MB
    int*    topk   = (int*)   (ws + 11010048);       //  1.0 MB
    float*  xo     = (float*) (ws + 12058624);       //  3.0 MB
    __bf16* qkvf   = (__bf16*)(ws + 15204352);       // 75.5 MB
    float*  yo     = (float*) (ws + 90701824);       // 50.3 MB (also t1)
    __bf16* z2_bf  = (__bf16*)(ws + 141033472);      // 25.2 MB
    __bf16* x_bf   = (__bf16*)(ws + 166199296);      // 25.2 MB
    __bf16* qkvw_b = (__bf16*)(ws + 191365120);      //  0.9 MB
    __bf16* pww_b  = (__bf16*)(ws + 192249856);      //  0.3 MB

    // 0. one-time bf16 conversions (operands for the WMMA GEMMs)
    cvt_bf16_kernel<<<(8*64*64*384 + 255)/256, 256, 0, stream>>>(x, x_bf, 8*64*64*384);
    cvt_bf16_kernel<<<(1152*384 + 255)/256, 256, 0, stream>>>(qkv_w, qkvw_b, 1152*384);
    cvt_bf16_kernel<<<(384*384 + 255)/256, 256, 0, stream>>>(pw_w, pww_b, 384*384);

    // 1. coarse tokens (bf16 out)
    avgpool_kernel<<<(8*256*384 + 255)/256, 256, 0, stream>>>(x, xc_bf);

    // 2. coarse QKV (fp32 out, scalar consumer)
    { int M = 2048, N = 1152, K = 384;
      int T = (M/16)*(N/64);
      gemm_bf16_wmma<<<(T+7)/8, 256, 0, stream>>>(xc_bf, qkvw_b, nullptr, qkvc, nullptr, M, N, K); }

    // 3. coarse attention + top-4 routing
    coarse_attn_kernel<<<(8*8*256 + 255)/256, 256, 0, stream>>>(qkvc, xo, topk);

    // 4. full-res QKV in bf16 (feeds WMMA fragments directly; fits in L2)
    { int M = 32768, N = 1152, K = 384;
      int T = (M/16)*(N/64);
      gemm_bf16_wmma<<<(T+7)/8, 256, 0, stream>>>(x_bf, qkvw_b, nullptr, nullptr, qkvf, M, N, K); }

    // 5. windowed gathered attention (one wave per (b,h,window))
    win_attn_kernel<<<(8*8*256)/8, 256, 0, stream>>>(qkvf, topk, yo);

    // 6. upsample+dwconv mix (in-place into yo), then second dwconv (bf16 out)
    upconv_mix_kernel<<<(8*64*64*384 + 255)/256, 256, 0, stream>>>(xo, dwx_w, dwx_b, yo);
    dwconv_kernel<<<(8*64*64*384 + 255)/256, 256, 0, stream>>>(yo, dw_w, dw_b, z2_bf);

    // 7. pointwise projection + bias -> output (B,H,W,C) f32
    { int M = 32768, N = 384, K = 384;
      int T = (M/16)*(N/64);
      gemm_bf16_wmma<<<(T+7)/8, 256, 0, stream>>>(z2_bf, pww_b, pw_b, (float*)d_out, nullptr, M, N, K); }
}
